// GeneralizedRingAttractorNoGain_77292231459485
// MI455X (gfx1250) — matrix operand order repaired and checked by reading, not compiled, not run on hardware
//
#include <hip/hip_runtime.h>
#include <hip/hip_bf16.h>
#include <math.h>
#include <stdint.h>

// Problem constants (from reference setup_inputs)
#define BB 64
#define TT 256
#define KK 32
#define NN 512
#define ROWS_CAT (KK * NN + NN)   // Wa rows (16384) + W0 rows (512) = 16896

typedef __attribute__((ext_vector_type(16))) _Float16 v16h;
typedef __attribute__((ext_vector_type(8)))  _Float16 v8h;
typedef __attribute__((ext_vector_type(8)))  float    v8f;

union AFrag { v16h v; struct { v8h lo, hi; } s; };

// ---------------------------------------------------------------------------
// WMMA fragment helpers (wave32, v_wmma_f32_16x16x32_f16)
// ---------------------------------------------------------------------------

// A fragment: 16x32 f16 tile at `base` (row-major, leading dim ld halfs).
// ISA 7.12.2 16-bit A 16x32: lanes 0-15 -> row M=lane, K halves {0..7,16..23};
// lanes 16-31 -> row M=lane-16, K halves {8..15,24..31}.
__device__ __forceinline__ v16h load_a_frag(const _Float16* base, int ld, int lane) {
    int m  = lane & 15;
    int kb = (lane < 16) ? 0 : 8;
    const _Float16* p = base + m * ld + kb;
    AFrag u;
    u.s.lo = *(const v8h*)(p);
    u.s.hi = *(const v8h*)(p + 16);
    return u.v;
}

// B fragment: 32x16 f16 tile, element (k, col) stored at colbase[col*ld + k]
// (i.e. column-major). Lanes 0-15 -> col=lane, K {0..7,16..23}; lanes 16-31 -> K {8..15,24..31}.
__device__ __forceinline__ v16h load_b_frag(const _Float16* colbase, int ld, int lane) {
    int c  = lane & 15;
    int kb = (lane < 16) ? 0 : 8;
    const _Float16* p = colbase + c * ld + kb;
    AFrag u;
    u.s.lo = *(const v8h*)(p);
    u.s.hi = *(const v8h*)(p + 16);
    return u.v;
}

// C/D fragment store: 16x16 f32 tile at `base` (row-major, leading dim ldc).
// Lanes 0-15: N=lane, VGPR g -> M=g. Lanes 16-31: N=lane-16, VGPR g -> M=8+g.
__device__ __forceinline__ void store_c_frag(float* base, int ldc, int lane, v8f c) {
    int n  = lane & 15;
    int mb = (lane < 16) ? 0 : 8;
    float* p = base + mb * ldc + n;
#pragma unroll
    for (int g = 0; g < 8; ++g) p[g * ldc] = c[g];
}

#define WMMA_F16(a, b, c) \
    __builtin_amdgcn_wmma_f32_16x16x32_f16(false, (a), false, (b), (short)0, (c), false, false)

// ---------------------------------------------------------------------------
// One-time prep kernels
// ---------------------------------------------------------------------------

// Build Acat_h (f16): rows [0,16384) = Wa[k,n,:]; rows [16384,16896) = W0 = -0.1 + 0.1*Wo
__global__ void __launch_bounds__(256) prep_acat_kernel(const float* Wa, const float* Wo,
                                                        _Float16* Acat) {
    int idx = blockIdx.x * 256 + threadIdx.x;       // 0 .. ROWS_CAT*NN-1
    int row = idx >> 9;                              // /512
    int m   = idx & (NN - 1);
    float val;
    if (row < KK * NN) val = Wa[idx];
    else               val = -0.1f + 0.1f * Wo[(row - KK * NN) * NN + m];
    Acat[idx] = (_Float16)val;
}

// W_delta7[n,m] = cos(2*pi*(n-m)/N); symmetric, so row-major == column-major.
__global__ void __launch_bounds__(256) prep_wd7_kernel(_Float16* Wd7h) {
    int idx = blockIdx.x * 256 + threadIdx.x;       // 0 .. NN*NN-1
    int n = idx >> 9;
    int m = idx & (NN - 1);
    float v = cosf(6.28318530717958647692f * (float)(n - m) / (float)NN);
    Wd7h[idx] = (_Float16)v;
}

// Initial bump: center = N/2, width = N/10, L2-normalized; identical per batch row.
__global__ void prep_r0_kernel(float* r32, _Float16* rh) {
    __shared__ float ssum[NN];
    int n = threadIdx.x;                             // 0..511
    float d = fabsf((float)n - (float)(NN / 2));
    d = fminf(d, (float)NN - d);
    const float bw = (float)NN / 10.0f;              // 51.2
    float bump = expf(-(d * d) / (2.0f * bw * bw));
    ssum[n] = bump * bump;
    __syncthreads();
    for (int s = NN / 2; s > 0; s >>= 1) {
        if (n < s) ssum[n] += ssum[n + s];
        __syncthreads();
    }
    float val = bump / sqrtf(ssum[0]);
    for (int b = 0; b < BB; ++b) {
        r32[b * NN + n] = val;
        rh[b * NN + n]  = (_Float16)val;
    }
}

// ---------------------------------------------------------------------------
// Per-step kernels
// ---------------------------------------------------------------------------

// vcat[row][b] = sum_m Acat[row][m] * r[b][m]   (row in [0,16896), b in [0,64))
// Block = 8 waves. r (64x512 f16 = 64KB) is staged once into LDS with
// CDNA5 async global->LDS copies (ASYNCcnt), then every wave's B fragments
// come from LDS while Wa fragments stream from L2, software-pipelined.
__global__ void __launch_bounds__(256) gemm_v_kernel(const _Float16* __restrict__ Acat,
                                                     const _Float16* __restrict__ rh,
                                                     float* __restrict__ vcat) {
    __shared__ __align__(32) _Float16 rsh[BB * NN];  // 64 KB
    int tid  = threadIdx.x;
    int wave = tid >> 5;
    int lane = tid & 31;

    // --- async stage r into LDS: 4096 x b128, 16 per thread ---------------
    for (int i = tid; i < (BB * NN) / 8; i += 256) {
        uint32_t ldsoff = (uint32_t)(uintptr_t)(&rsh[i * 8]);
        uint64_t gaddr  = (uint64_t)(uintptr_t)(rh + i * 8);
        asm volatile("global_load_async_to_lds_b128 %0, %1, off"
                     :: "v"(ldsoff), "v"(gaddr) : "memory");
    }
    asm volatile("s_wait_asynccnt 0x0" ::: "memory");
    __syncthreads();

    // --- WMMA main loop ----------------------------------------------------
    int strip = blockIdx.x * 8 + wave;               // 0..1055
    const _Float16* ap = Acat + (size_t)(strip * 16) * NN
                              + (lane & 15) * NN + ((lane < 16) ? 0 : 8);

    v8f c0 = {}, c1 = {}, c2 = {}, c3 = {};
    v8h alo = *(const v8h*)(ap);
    v8h ahi = *(const v8h*)(ap + 16);

#pragma unroll 4
    for (int kk = 0; kk < NN; kk += 32) {
        AFrag u; u.s.lo = alo; u.s.hi = ahi;
        v16h a = u.v;
        if (kk + 32 < NN) {                          // prefetch next A fragment
            alo = *(const v8h*)(ap + kk + 32);
            ahi = *(const v8h*)(ap + kk + 48);
        }
        v16h b0 = load_b_frag(rsh +  0 * NN + kk, NN, lane);
        v16h b1 = load_b_frag(rsh + 16 * NN + kk, NN, lane);
        v16h b2 = load_b_frag(rsh + 32 * NN + kk, NN, lane);
        v16h b3 = load_b_frag(rsh + 48 * NN + kk, NN, lane);
        c0 = WMMA_F16(a, b0, c0);
        c1 = WMMA_F16(a, b1, c1);
        c2 = WMMA_F16(a, b2, c2);
        c3 = WMMA_F16(a, b3, c3);
    }
    int row0 = strip * 16;
    store_c_frag(vcat + row0 * BB +  0, BB, lane, c0);
    store_c_frag(vcat + row0 * BB + 16, BB, lane, c1);
    store_c_frag(vcat + row0 * BB + 32, BB, lane, c2);
    store_c_frag(vcat + row0 * BB + 48, BB, lane, c3);
}

// rec[b,n] = vcat[W0 row] + sum_k A_t[b,k]*vcat[k*N+n]; tanh; EMA; emit r32/rh/bump.
__global__ void __launch_bounds__(256) combine_kernel(const float* __restrict__ vcat,
                                                      const float* __restrict__ action,
                                                      float* __restrict__ r32,
                                                      _Float16* __restrict__ rh,
                                                      float* __restrict__ bump_out,
                                                      int t) {
    int idx = blockIdx.x * 256 + threadIdx.x;        // 0 .. B*N-1
    int b = idx & (BB - 1);
    int n = idx >> 6;
    float rec = vcat[(KK * NN + n) * BB + b];        // W0 @ r part
    const float* arow = action + (b * TT + t) * KK;
#pragma unroll
    for (int k = 0; k < KK; ++k)
        rec += arow[k] * vcat[(k * NN + n) * BB + b];
    rec = tanhf(rec);
    float rold = r32[b * NN + n];
    float rnew = rold * 0.85f + 0.15f * rec;         // ALPHA = 0.15
    r32[b * NN + n] = rnew;
    rh[b * NN + n]  = (_Float16)rnew;
    bump_out[(b * TT + t) * NN + n] = rnew;
}

// rd7raw[b][n] = sum_m r[b][m] * Wd7[m][n]; Wd7 symmetric -> row-major works as B.
// 128 tiles (4 row strips x 32 col tiles); 8 waves/block x 16 blocks.
__global__ void __launch_bounds__(256) gemm_rd7_kernel(const _Float16* __restrict__ rh,
                                                       const _Float16* __restrict__ Wd7h,
                                                       float* __restrict__ rd7raw) {
    int wave = threadIdx.x >> 5;
    int lane = threadIdx.x & 31;
    int tile = blockIdx.x * 8 + wave;                // 0..127
    int row0 = (tile & 3) * 16;                      // b-strip
    int col0 = (tile >> 2) * 16;                     // n-tile
    v8f c = {};
#pragma unroll 4
    for (int kk = 0; kk < NN; kk += 32) {
        v16h a = load_a_frag(rh + row0 * NN + kk, NN, lane);
        // element (m, n) of Wd7 at Wd7h[n*NN + m] (symmetry)
        v16h b = load_b_frag(Wd7h + col0 * NN + kk, NN, lane);
        c = WMMA_F16(a, b, c);
    }
    store_c_frag(rd7raw + row0 * NN + col0, NN, lane, c);
}

// Row-max normalize: rd7_out[b,t,n] = rd7raw[b,n] / max_n rd7raw[b,n]
__global__ void __launch_bounds__(256) norm_kernel(const float* __restrict__ rd7raw,
                                                   float* __restrict__ rd7_out, int t) {
    __shared__ float smax[256];
    int b = blockIdx.x;
    int tid = threadIdx.x;
    float v0 = rd7raw[b * NN + tid];
    float v1 = rd7raw[b * NN + tid + 256];
    smax[tid] = fmaxf(v0, v1);
    __syncthreads();
    for (int s = 128; s > 0; s >>= 1) {
        if (tid < s) smax[tid] = fmaxf(smax[tid], smax[tid + s]);
        __syncthreads();
    }
    float m = smax[0];
    float* o = rd7_out + (b * TT + t) * NN;
    o[tid]       = v0 / m;
    o[tid + 256] = v1 / m;
}

// ---------------------------------------------------------------------------
// Host launch
// ---------------------------------------------------------------------------
extern "C" void kernel_launch(void* const* d_in, const int* in_sizes, int n_in,
                              void* d_out, int out_size, void* d_ws, size_t ws_size,
                              hipStream_t stream) {
    const float* action = (const float*)d_in[0];     // (B,T,K)
    const float* Wo     = (const float*)d_in[1];     // (N,N)
    const float* Wa     = (const float*)d_in[2];     // (K,N,N)

    float* rd7_out  = (float*)d_out;                          // (B,T,N)
    float* bump_out = (float*)d_out + (size_t)BB * TT * NN;   // (B,T,N)

    // Workspace carve-up (f16 region first, then f32 region; all 16B aligned)
    _Float16* Acat = (_Float16*)d_ws;                         // 16896*512 halfs
    _Float16* Wd7h = Acat + (size_t)ROWS_CAT * NN;            // 512*512 halfs
    _Float16* rh   = Wd7h + (size_t)NN * NN;                  // 64*512 halfs
    float* r32     = (float*)(rh + (size_t)BB * NN);          // 64*512 f32
    float* vcat    = r32 + (size_t)BB * NN;                   // 16896*64 f32
    float* rd7raw  = vcat + (size_t)ROWS_CAT * BB;            // 64*512 f32

    // One-time prep
    prep_acat_kernel<<<(ROWS_CAT * NN) / 256, 256, 0, stream>>>(Wa, Wo, Acat);
    prep_wd7_kernel<<<(NN * NN) / 256, 256, 0, stream>>>(Wd7h);
    prep_r0_kernel<<<1, NN, 0, stream>>>(r32, rh);

    // Recurrent time loop: 4 kernels per step, all dependency-ordered on stream.
    for (int t = 0; t < TT; ++t) {
        gemm_v_kernel<<<(ROWS_CAT / 16) / 8, 256, 0, stream>>>(Acat, rh, vcat);
        combine_kernel<<<(BB * NN) / 256, 256, 0, stream>>>(vcat, action, r32, rh, bump_out, t);
        gemm_rd7_kernel<<<((BB / 16) * (NN / 16)) / 8, 256, 0, stream>>>(rh, Wd7h, rd7raw);
        norm_kernel<<<BB, 256, 0, stream>>>(rd7raw, rd7_out, t);
    }
}
